// SubjectLayers_22454089023725
// MI455X (gfx1250) — matrix operand order repaired and checked by reading, not compiled, not run on hardware
//
#include <hip/hip_runtime.h>
#include <cstddef>

// CDNA5 (gfx1250) SubjectLayers: out[b] = W[subjects[b]]^T @ x[b]
//   x: (32, 4096, 256) fp32   weights: (8, 4096, 1024) fp32   out: (32, 1024, 256) fp32
//
// FP32-accurate GEMM on the bf16 matrix pipe: each fp32 operand is split into
// hi/lo bf16 (a = ah + al exactly to ~2^-17 rel), and each 16x16x32 product is
// computed as ah*bh + ah*bl + al*bh with three chained V_WMMA_F32_16X16X32_BF16.
// 3 bf16 WMMAs replace eight K-deep V_WMMA_F32_16X16X4_F32 -> ~2.7x matrix-pipe
// throughput at ~1e-5 relative accuracy (vs ~240 FLOP/B arithmetic intensity,
// this problem is compute-bound, so the matrix pipe choice dominates).

typedef __attribute__((ext_vector_type(8)))  float  v8f;
typedef __attribute__((ext_vector_type(8)))  __bf16 v8bf;
typedef __attribute__((ext_vector_type(16))) __bf16 v16bf;

static constexpr int kC = 4096;  // IN_CH  (GEMM K)
static constexpr int kD = 1024;  // OUT_CH (GEMM M)
static constexpr int kT = 256;   // TIME   (GEMM N)
static constexpr int kB = 32;    // BATCH
static constexpr int kNS = 8;    // N_SUBJECTS

static constexpr int BM = 128;        // block tile in D
static constexpr int BN = 128;        // block tile in T
static constexpr int BK = 32;         // K per stage = one bf16 WMMA K-depth
static constexpr int KP = BK + 8;     // padded LDS row length (bf16 elements)
static constexpr int NST = kC / BK;   // 128 stages

#define WMMA_BF16(A_, B_, C_)                                            \
  __builtin_amdgcn_wmma_f32_16x16x32_bf16(false, (A_), false, (B_),      \
                                          (short)0, (C_), false, false)

__device__ __forceinline__ v16bf cat8(v8bf a, v8bf b) {
  return __builtin_shufflevector(a, b, 0, 1, 2, 3, 4, 5, 6, 7,
                                 8, 9, 10, 11, 12, 13, 14, 15);
}

__global__ __launch_bounds__(256, 1)
void subject_layers_bf16x3(const float* __restrict__ x,
                           const int* __restrict__ subjects,
                           const float* __restrict__ w,
                           float* __restrict__ out) {
  // hi/lo bf16 operand tiles, [row][k] with k contiguous so WMMA fragments are
  // two aligned ds_load_b128 each. 4 * 128 * 40 * 2B = 40 KB.
  __shared__ alignas(16) __bf16 Ah[BM * KP];
  __shared__ alignas(16) __bf16 Al[BM * KP];
  __shared__ alignas(16) __bf16 Bh[BN * KP];
  __shared__ alignas(16) __bf16 Bl[BN * KP];

  const int b    = blockIdx.x;
  const int m0   = blockIdx.y * BM;   // D-block origin
  const int n0   = blockIdx.z * BN;   // T-block origin
  const int subj = subjects[b] & (kNS - 1);

  const float* __restrict__ wg = w + (size_t)subj * kC * kD;  // [c][d]
  const float* __restrict__ xg = x + (size_t)b * kC * kT;     // [c][t]

  const int tid  = threadIdx.x;
  const int lane = tid & 31;
  const int wid  = tid >> 5;            // 0..7
  const int waveM = (wid & 1) * 64;     // 2 waves tile M -> 4 m-tiles each
  const int waveN = (wid >> 1) * 32;    // 4 waves tile N -> 2 n-tiles each
  const int half = (lane >> 4) & 1;     // K-range selector per ISA 16-bit layout
  const int l16  = lane & 15;

  // LDS fill ownership: row r (m for A, n for B), K segment of 16.
  const int r   = tid & 127;
  const int seg = tid >> 7;             // 0 or 1

  v8f acc[4][2];
#pragma unroll
  for (int mi = 0; mi < 4; ++mi)
#pragma unroll
    for (int ni = 0; ni < 2; ++ni)
#pragma unroll
      for (int e = 0; e < 8; ++e) acc[mi][ni][e] = 0.0f;

  // Register staging for the next K-stage (global loads overlap compute).
  float ra[16], rb[16];

  auto load_regs = [&](int k0) {
    const float* pA = wg + (size_t)(k0 + seg * 16) * kD + (m0 + r);
    const float* pB = xg + (size_t)(k0 + seg * 16) * kT + (n0 + r);
#pragma unroll
    for (int j = 0; j < 16; ++j) {
      ra[j] = pA[(size_t)j * kD];   // coalesced: lanes span consecutive d
      rb[j] = pB[(size_t)j * kT];   // coalesced: lanes span consecutive t
    }
  };

  // Split fp32 -> (hi, lo) bf16 and store one 16-wide K run, [row][k] layout.
  auto split_store = [&](float (&vv)[16], __bf16* H, __bf16* L) {
    v8bf h0, h1, l0, l1;
#pragma unroll
    for (int j = 0; j < 8; ++j) {
      float v  = vv[j];
      __bf16 h = (__bf16)v;
      l0[j]    = (__bf16)(v - (float)h);
      h0[j]    = h;
      float v2  = vv[j + 8];
      __bf16 h2 = (__bf16)v2;
      l1[j]     = (__bf16)(v2 - (float)h2);
      h1[j]     = h2;
    }
    __bf16* ph = H + r * KP + seg * 16;
    *reinterpret_cast<v8bf*>(ph)     = h0;
    *reinterpret_cast<v8bf*>(ph + 8) = h1;
    __bf16* pl = L + r * KP + seg * 16;
    *reinterpret_cast<v8bf*>(pl)     = l0;
    *reinterpret_cast<v8bf*>(pl + 8) = l1;
  };

  // A fragment (16x32 bf16): lane m; elems 0-15 = K 16*half .. +15.
  auto loadA = [&](const __bf16* base, int mi) -> v16bf {
    const __bf16* p = base + (waveM + mi * 16 + l16) * KP + 16 * half;
    v8bf x0 = *reinterpret_cast<const v8bf*>(p);
    v8bf x1 = *reinterpret_cast<const v8bf*>(p + 8);
    return cat8(x0, x1);
  };
  // B fragment (32x16 bf16): lane n; elems 0-15 = K 16*half .. +15.
  auto loadB = [&](const __bf16* base, int ni) -> v16bf {
    const __bf16* p = base + (waveN + ni * 16 + l16) * KP + 16 * half;
    v8bf x0 = *reinterpret_cast<const v8bf*>(p);
    v8bf x1 = *reinterpret_cast<const v8bf*>(p + 8);
    return cat8(x0, x1);
  };

  load_regs(0);

  for (int st = 0; st < NST; ++st) {
    __syncthreads();                 // previous compute finished reading LDS
    split_store(ra, Ah, Al);
    split_store(rb, Bh, Bl);
    __syncthreads();

    if (st + 1 < NST) load_regs((st + 1) * BK);  // HBM latency hides under WMMA

    v16bf bhf[2], blf[2];
#pragma unroll
    for (int ni = 0; ni < 2; ++ni) {
      bhf[ni] = loadB(Bh, ni);
      blf[ni] = loadB(Bl, ni);
    }
#pragma unroll
    for (int mi = 0; mi < 4; ++mi) {
      v16bf ah = loadA(Ah, mi);
      v16bf al = loadA(Al, mi);
#pragma unroll
      for (int ni = 0; ni < 2; ++ni) {
        // small terms first, then dominant: acc += al*bh + ah*bl + ah*bh
        acc[mi][ni] = WMMA_BF16(al, bhf[ni], acc[mi][ni]);
        acc[mi][ni] = WMMA_BF16(ah, blf[ni], acc[mi][ni]);
        acc[mi][ni] = WMMA_BF16(ah, bhf[ni], acc[mi][ni]);
      }
    }
  }

  // C/D layout: VGPR e, lanes 0-15 -> M=e, lanes 16-31 -> M=e+8; N = lane&15.
  float* __restrict__ og = out + (size_t)b * kD * kT;
#pragma unroll
  for (int mi = 0; mi < 4; ++mi) {
#pragma unroll
    for (int ni = 0; ni < 2; ++ni) {
      const int d0 = m0 + waveM + mi * 16 + 8 * half;
      const int t0 = n0 + waveN + ni * 16 + l16;
#pragma unroll
      for (int e = 0; e < 8; ++e)
        og[(size_t)(d0 + e) * kT + t0] = acc[mi][ni][e];
    }
  }
}

extern "C" void kernel_launch(void* const* d_in, const int* in_sizes, int n_in,
                              void* d_out, int out_size, void* d_ws, size_t ws_size,
                              hipStream_t stream) {
  (void)in_sizes; (void)n_in; (void)out_size; (void)d_ws; (void)ws_size;
  const float* x        = (const float*)d_in[0];
  const int*   subjects = (const int*)d_in[1];
  const float* w        = (const float*)d_in[2];
  float*       out      = (float*)d_out;

  dim3 grid(kB, kD / BM, kT / BN);  // (32, 8, 2)
  subject_layers_bf16x3<<<grid, dim3(256), 0, stream>>>(x, subjects, w, out);
}
